// RoutedEdgeClassifier_75617194213651
// MI455X (gfx1250) — matrix operand order repaired and checked by reading, not compiled, not run on hardware
//
#include <hip/hip_runtime.h>
#include <hip/hip_bf16.h>

#define DEV static __device__ __forceinline__

typedef __attribute__((ext_vector_type(16))) _Float16 v16h;
typedef __attribute__((ext_vector_type(8)))  _Float16 v8h;
typedef __attribute__((ext_vector_type(8)))  float    v8f;

constexpr int NN = 4096;     // nodes
constexpr int NE = 8192;     // edges
constexpr int D  = 128;      // feature dim
constexpr int KN = 1638;     // int(0.4*4096)
constexpr int KE = 3276;     // int(0.4*8192)
constexpr float QSCALE = 0.17677669529663687f;  // 1/sqrt(32)

// ---------------- helpers ----------------

DEV v8f zero8() { v8f z;
#pragma unroll
  for (int i = 0; i < 8; ++i) z[i] = 0.f;
  return z; }

// A-fragment (16x32 f16): lane holds row (lane&15); halves h -> k = h + 16*hi.
// One contiguous 16-half chunk, loaded as two 8-half pieces.
DEV v16h load_afrag(const _Float16* rowbase, int hi) {
  v8h lo = *(const v8h*)(rowbase + 16 * hi);
  v8h up = *(const v8h*)(rowbase + 16 * hi + 8);
  v16h r;
#pragma unroll
  for (int i = 0; i < 8; ++i) { r[i] = lo[i]; r[i + 8] = up[i]; }
  return r;
}

// B-fragment (32x16 f16): lane holds col (lane&15); halves h -> k = h + 16*hi.
// When the 16 halves are contiguous in memory (per-lane), just load 2x b128.
DEV v16h load_h16(const _Float16* p) {
  v8h lo = *(const v8h*)(p);
  v8h up = *(const v8h*)(p + 8);
  v16h r;
#pragma unroll
  for (int i = 0; i < 8; ++i) { r[i] = lo[i]; r[i + 8] = up[i]; }
  return r;
}

DEV v8f wmma16(v16h a, v16h b, v8f c) {
  return __builtin_amdgcn_wmma_f32_16x16x32_f16(false, a, false, b, (short)0, c,
                                                false, false);
}

DEV unsigned mapf(float f) {   // monotone float -> uint (ascending)
  unsigned u = __float_as_uint(f);
  return (u & 0x80000000u) ? ~u : (u | 0x80000000u);
}

DEV float gelu_tanh(float x) {
  float x3 = x * x * x;
  return 0.5f * x * (1.f + tanhf(0.79788456080286535588f * (x + 0.044715f * x3)));
}

// ---------------- 1) router scores ----------------
__global__ void scores_kernel(const float* __restrict__ nf, const float* __restrict__ ef,
                              const float* __restrict__ Wn, const float* __restrict__ bn,
                              const float* __restrict__ We, const float* __restrict__ be,
                              float* __restrict__ ns, float* __restrict__ es) {
  int wave = threadIdx.x >> 5, lane = threadIdx.x & 31;
  int row = blockIdx.x * 8 + wave;
  const float* feat; const float* w; float b; float* out;
  if (row < NN)            { feat = nf + (size_t)row * D;        w = Wn; b = bn[0]; out = ns + row; }
  else if (row < NN + NE)  { int r = row - NN; feat = ef + (size_t)r * D; w = We; b = be[0]; out = es + r; }
  else return;
  float4 f = ((const float4*)feat)[lane];
  float4 v = ((const float4*)w)[lane];
  float p = f.x * v.x + f.y * v.y + f.z * v.z + f.w * v.w;
#pragma unroll
  for (int off = 16; off > 0; off >>= 1) p += __shfl_xor(p, off, 32);
  if (lane == 0) *out = p + b;
}

// ---------------- 2) exact k-th largest via 4-pass radix select ----------------
__global__ void select_kernel(const float* __restrict__ ns, const float* __restrict__ es,
                              unsigned* __restrict__ thr) {
  __shared__ unsigned hist[256];
  __shared__ unsigned s_prefix;
  __shared__ int s_k;
  const float* data = (blockIdx.x == 0) ? ns : es;
  int n  = (blockIdx.x == 0) ? NN : NE;
  int k0 = (blockIdx.x == 0) ? KN : KE;
  int tid = threadIdx.x;
  if (tid == 0) { s_prefix = 0u; s_k = k0; }
  __syncthreads();
  for (int pass = 0; pass < 4; ++pass) {
    hist[tid] = 0u;
    __syncthreads();
    unsigned pref = s_prefix;
    int shift = 24 - 8 * pass;
    for (int i = tid; i < n; i += 256) {
      unsigned key = mapf(data[i]);
      if (pass == 0 || (key >> (shift + 8)) == pref)
        atomicAdd(&hist[(key >> shift) & 255u], 1u);
    }
    __syncthreads();
    if (tid == 0) {
      unsigned cum = 0; int b = 255;
      for (; b >= 0; --b) { cum += hist[b]; if ((int)cum >= s_k) break; }
      if (b < 0) b = 0;
      s_k -= (int)(cum - hist[b]);
      s_prefix = (pref << 8) | (unsigned)b;
    }
    __syncthreads();
  }
  if (tid == 0) thr[blockIdx.x] = s_prefix;
}

// ---------------- 3) masks + weighted (f32 -> f16) ----------------
__global__ void nmask_kernel(const float* __restrict__ ns, const unsigned* __restrict__ thr,
                             unsigned char* __restrict__ nmask) {
  int i = blockIdx.x * 256 + threadIdx.x;
  if (i < NN) nmask[i] = (mapf(ns[i]) >= thr[0]) ? 1 : 0;
}

__global__ void weighted_kernel(const float* __restrict__ ef, const float* __restrict__ es,
                                const unsigned* __restrict__ thr,
                                const unsigned char* __restrict__ nmask,
                                const int* __restrict__ ei, _Float16* __restrict__ wh) {
  int idx = blockIdx.x * 256 + threadIdx.x;   // < NE*D
  int e = idx >> 7;
  bool m = (mapf(es[e]) >= thr[1]) && nmask[ei[e]] && nmask[ei[NE + e]];
  wh[idx] = m ? (_Float16)ef[idx] : (_Float16)0.f;
}

// ---------------- 4) pack weights into per-lane B-fragment layout ----------------
// bpack halves layout per matrix: [ct][kk][lane][h], element = W[kk*32 + h + 16*hi][ct*16 + (lane&15)]
__global__ void packw_kernel(const float* __restrict__ Wq, const float* __restrict__ Wk,
                             const float* __restrict__ Wv, const float* __restrict__ Wo,
                             const float* __restrict__ W1, const float* __restrict__ W2,
                             const float* __restrict__ bq, _Float16* __restrict__ bpack,
                             float* __restrict__ bqs) {
  int idx = blockIdx.x * 256 + threadIdx.x;
  if (idx < 5 * 16384) {
    int w = idx / 16384, r = idx % 16384;
    int kk = (r >> 9) & 3, lane = (r >> 4) & 31, h = r & 15;
    int cn = lane & 15, hi = lane >> 4;
    int krow = kk * 32 + h + 16 * hi;
    int col = ((r >> 11) << 4) + cn;            // ct*16 + n
    const float* W = (w == 0) ? Wq : (w == 1) ? Wk : (w == 2) ? Wv : (w == 3) ? Wo : W1;
    float val = W[krow * D + col];
    if (w == 0) val *= QSCALE;                  // fold 1/sqrt(dh) into Wq
    bpack[idx] = (_Float16)val;
  } else if (idx < 5 * 16384 + 2048) {          // W2 padded 128x16
    int r = idx - 5 * 16384;
    int kk = (r >> 9) & 3, lane = (r >> 4) & 31, h = r & 15;
    int cn = lane & 15, hi = lane >> 4;
    int krow = kk * 32 + h + 16 * hi;
    bpack[idx] = (_Float16)((cn < 10) ? W2[krow * 10 + cn] : 0.f);
  } else if (idx < 5 * 16384 + 2048 + 128) {
    int i = idx - (5 * 16384 + 2048);
    bqs[i] = bq[i] * QSCALE;                    // fold 1/sqrt(dh) into bq
  }
}

// ---------------- 5a) fused Q/K/V projection: A-fragments loaded once, 3 WMMA chains ------
__global__ void qkv_gemm_kernel(const _Float16* __restrict__ X, const _Float16* __restrict__ Bp,
                                const float* __restrict__ bq, const float* __restrict__ bk,
                                const float* __restrict__ bv,
                                _Float16* __restrict__ qh, _Float16* __restrict__ kh,
                                _Float16* __restrict__ vh) {
  int wave = threadIdx.x >> 5, lane = threadIdx.x & 31;
  int cn = lane & 15, hi = lane >> 4;
  int gw = blockIdx.x * 8 + wave;
  int rt = gw >> 3, ct = gw & 7;
  const _Float16* xrow = X + (size_t)(rt * 16 + cn) * D;
  v8f aq = zero8(), ak = zero8(), av = zero8();
#pragma unroll
  for (int kk = 0; kk < 4; ++kk) {
    v16h A  = load_afrag(xrow + kk * 32, hi);
    size_t bo_ = ((size_t)(ct * 4 + kk) * 32 + lane) * 16;
    v16h Bq_ = load_h16(Bp + bo_);
    v16h Bk_ = load_h16(Bp + 16384 + bo_);
    v16h Bv_ = load_h16(Bp + 32768 + bo_);
    aq = wmma16(A, Bq_, aq);
    ak = wmma16(A, Bk_, ak);
    av = wmma16(A, Bv_, av);
  }
  int col = ct * 16 + cn;
  float bvq = bq[col], bvk = bk[col], bvv = bv[col];
#pragma unroll
  for (int g = 0; g < 8; ++g) {
    size_t row = (size_t)(rt * 16 + g + 8 * hi);
    qh[row * D + col] = (_Float16)(aq[g] + bvq);
    kh[row * D + col] = (_Float16)(ak[g] + bvk);
    vh[row * D + col] = (_Float16)(av[g] + bvv);
  }
}

// ---------------- 5b) generic WMMA GEMM: Y[8192 x nct*16] = X[8192x128] @ Bpack + bias ------
// mode 0: f16 row-major out (ld=128); mode 1: gelu then f16; mode 2: f32 out, cols<10 -> Yf (ld=10)
__global__ void gemm_kernel(const _Float16* __restrict__ X, const _Float16* __restrict__ Bp,
                            const float* __restrict__ bias, int bias_n, int nct, int mode,
                            _Float16* __restrict__ Yh, float* __restrict__ Yf) {
  int wave = threadIdx.x >> 5, lane = threadIdx.x & 31;
  int cn = lane & 15, hi = lane >> 4;
  int gw = blockIdx.x * 8 + wave;
  int rt = gw / nct, ct = gw % nct;
  if (rt >= NE / 16) return;
  const _Float16* xrow = X + (size_t)(rt * 16 + cn) * D;
  v8f acc = zero8();
#pragma unroll
  for (int kk = 0; kk < 4; ++kk) {
    v16h A = load_afrag(xrow + kk * 32, hi);
    v16h B = load_h16(Bp + ((size_t)(ct * 4 + kk) * 32 + lane) * 16);
    acc = wmma16(A, B, acc);
  }
  int col = ct * 16 + cn;
  float bv = (col < bias_n) ? bias[col] : 0.f;
#pragma unroll
  for (int g = 0; g < 8; ++g) {
    float v = acc[g] + bv;
    if (mode == 1) v = gelu_tanh(v);
    int row = rt * 16 + g + 8 * hi;
    if (mode == 2) { if (col < 10) Yf[(size_t)row * 10 + col] = v; }
    else Yh[(size_t)row * D + col] = (_Float16)v;
  }
}

// ---------------- 6) pack V into B-fragment layout for the PV wmma ----------------
// vpack[head][kb(256)][c(2)][lane][h] = V[kb*32 + h + 16*hi][head*32 + c*16 + (lane&15)]
__global__ void vpack_kernel(const _Float16* __restrict__ vh, _Float16* __restrict__ vpack) {
  int idx = blockIdx.x * 256 + threadIdx.x;      // < 4*256*2*32*16
  int h = idx & 15, lane = (idx >> 4) & 31, c = (idx >> 9) & 1;
  int kb = (idx >> 10) & 255, head = idx >> 18;
  int key = kb * 32 + h + 16 * (lane >> 4);
  int col = head * 32 + c * 16 + (lane & 15);
  vpack[idx] = vh[(size_t)key * D + col];
}

// ---------------- 7) flash attention with on-the-fly adjacency mask ----------------
__global__ void attn_kernel(const _Float16* __restrict__ qh, const _Float16* __restrict__ kh,
                            const _Float16* __restrict__ vpack, const int* __restrict__ ei,
                            _Float16* __restrict__ out) {
  __shared__ __align__(16) _Float16 Pl[8][16][40];   // P bounce, 40-half row pad
  int wave = threadIdx.x >> 5, lane = threadIdx.x & 31;
  int cn = lane & 15, hi = lane >> 4;
  int gw = blockIdx.x * 8 + wave;
  int head = gw & 3, qt = gw >> 2;                   // qt in [0,512)
  const int* src = ei; const int* dst = ei + NE;

  // Q fragment (scale already folded into Wq/bq)
  v16h A = load_afrag(qh + (size_t)(qt * 16 + cn) * D + head * 32, hi);

  int qs[8], qd[8];
#pragma unroll
  for (int g = 0; g < 8; ++g) {
    int qi = qt * 16 + g + 8 * hi;
    qs[g] = src[qi]; qd[g] = dst[qi];
  }

  float m[8], l[8];
#pragma unroll
  for (int g = 0; g < 8; ++g) { m[g] = -1e30f; l[g] = 0.f; }
  v8f O0 = zero8(), O1 = zero8();

  for (int kb = 0; kb < NE / 32; ++kb) {
    int key0 = kb * 32 + cn, key1 = kb * 32 + 16 + cn;
    v16h B0 = load_h16(kh + (size_t)key0 * D + head * 32 + 16 * hi);
    v16h B1 = load_h16(kh + (size_t)key1 * D + head * 32 + 16 * hi);
    // prefetch next key block (K rows + V fragments) into cache while softmax runs
    if (kb + 1 < NE / 32) {
      __builtin_prefetch((const void*)(kh + (size_t)(key0 + 32) * D + head * 32 + 16 * hi), 0, 1);
      __builtin_prefetch((const void*)(kh + (size_t)(key1 + 32) * D + head * 32 + 16 * hi), 0, 1);
      __builtin_prefetch((const void*)(vpack + (((size_t)(head * 256 + kb + 1) * 2) * 32 + lane) * 16), 0, 1);
    }
    v8f S0 = wmma16(A, B0, zero8());
    v8f S1 = wmma16(A, B1, zero8());

    int ks0 = src[key0], kd0 = dst[key0], ks1 = src[key1], kd1 = dst[key1];
    float s0[8], s1[8], mn[8];
#pragma unroll
    for (int g = 0; g < 8; ++g) {
      int qi = qt * 16 + g + 8 * hi;
      bool a0 = (qs[g] == ks0 || qs[g] == kd0 || qd[g] == ks0 || qd[g] == kd0) && (qi != key0);
      bool a1 = (qs[g] == ks1 || qs[g] == kd1 || qd[g] == ks1 || qd[g] == kd1) && (qi != key1);
      s0[g] = a0 ? S0[g] : -1e9f;
      s1[g] = a1 ? S1[g] : -1e9f;
    }
#pragma unroll
    for (int g = 0; g < 8; ++g) {                      // row max over 32 keys
      float v = fmaxf(s0[g], s1[g]);
      v = fmaxf(v, __shfl_xor(v, 1, 32));
      v = fmaxf(v, __shfl_xor(v, 2, 32));
      v = fmaxf(v, __shfl_xor(v, 4, 32));
      v = fmaxf(v, __shfl_xor(v, 8, 32));
      mn[g] = fmaxf(m[g], v);
    }
    float p0[8], p1[8];
#pragma unroll
    for (int g = 0; g < 8; ++g) {                      // rescale + exp + row sum
      float sc = __expf(m[g] - mn[g]);
      O0[g] *= sc; O1[g] *= sc; l[g] *= sc; m[g] = mn[g];
      p0[g] = __expf(s0[g] - mn[g]);
      p1[g] = __expf(s1[g] - mn[g]);
      float w = p0[g] + p1[g];
      w += __shfl_xor(w, 1, 32);
      w += __shfl_xor(w, 2, 32);
      w += __shfl_xor(w, 4, 32);
      w += __shfl_xor(w, 8, 32);
      l[g] += w;
    }
    // C-layout P -> A-fragment via LDS bounce (wave-private region)
#pragma unroll
    for (int g = 0; g < 8; ++g) {
      Pl[wave][g + 8 * hi][cn]      = (_Float16)p0[g];
      Pl[wave][g + 8 * hi][16 + cn] = (_Float16)p1[g];
    }
    asm volatile("s_wait_dscnt 0" ::: "memory");
    v16h Pa = load_afrag(&Pl[wave][cn][0], hi);

    v16h V0 = load_h16(vpack + (((size_t)(head * 256 + kb) * 2 + 0) * 32 + lane) * 16);
    v16h V1 = load_h16(vpack + (((size_t)(head * 256 + kb) * 2 + 1) * 32 + lane) * 16);
    O0 = wmma16(Pa, V0, O0);
    O1 = wmma16(Pa, V1, O1);
  }

#pragma unroll
  for (int g = 0; g < 8; ++g) {
    float inv = 1.f / l[g];
    int row = qt * 16 + g + 8 * hi;
    out[(size_t)row * D + head * 32 + cn]      = (_Float16)(O0[g] * inv);
    out[(size_t)row * D + head * 32 + 16 + cn] = (_Float16)(O1[g] * inv);
  }
}

// ---------------- host launch ----------------
extern "C" void kernel_launch(void* const* d_in, const int* in_sizes, int n_in,
                              void* d_out, int out_size, void* d_ws, size_t ws_size,
                              hipStream_t stream) {
  const float* node_f = (const float*)d_in[0];
  const float* edge_f = (const float*)d_in[1];
  const int*   ei     = (const int*)d_in[2];
  const float* Wn_r   = (const float*)d_in[3];
  const float* bn_r   = (const float*)d_in[4];
  const float* We_r   = (const float*)d_in[5];
  const float* be_r   = (const float*)d_in[6];
  const float* Wq = (const float*)d_in[7];  const float* bq = (const float*)d_in[8];
  const float* Wk = (const float*)d_in[9];  const float* bk = (const float*)d_in[10];
  const float* Wv = (const float*)d_in[11]; const float* bv = (const float*)d_in[12];
  const float* Wo = (const float*)d_in[13]; const float* bo = (const float*)d_in[14];
  const float* W1 = (const float*)d_in[15]; const float* b1 = (const float*)d_in[16];
  const float* W2 = (const float*)d_in[17]; const float* b2 = (const float*)d_in[18];
  float* logits = (float*)d_out;

  char* ws = (char*)d_ws;
  size_t off = 0;
  auto take = [&](size_t bytes) { size_t o = off; off = (off + bytes + 255) & ~(size_t)255; return o; };
  float*         ns     = (float*)(ws + take(NN * 4));
  float*         es     = (float*)(ws + take(NE * 4));
  unsigned*      thr    = (unsigned*)(ws + take(2 * 4));
  unsigned char* nmask  = (unsigned char*)(ws + take(NN));
  _Float16*      wgt    = (_Float16*)(ws + take((size_t)NE * D * 2));
  _Float16*      q_h    = (_Float16*)(ws + take((size_t)NE * D * 2));
  _Float16*      k_h    = (_Float16*)(ws + take((size_t)NE * D * 2));
  _Float16*      v_h    = (_Float16*)(ws + take((size_t)NE * D * 2));
  _Float16*      bpack  = (_Float16*)(ws + take((size_t)(5 * 16384 + 2048) * 2));
  float*         bqs    = (float*)(ws + take(128 * 4));
  _Float16*      vpk    = (_Float16*)(ws + take((size_t)4 * 256 * 2 * 32 * 16 * 2));
  _Float16*      attn   = (_Float16*)(ws + take((size_t)NE * D * 2));
  _Float16*      o_h    = (_Float16*)(ws + take((size_t)NE * D * 2));
  _Float16*      h1     = (_Float16*)(ws + take((size_t)NE * D * 2));
  (void)ws_size; (void)in_sizes; (void)n_in; (void)out_size;

  scores_kernel<<<(NN + NE) / 8, 256, 0, stream>>>(node_f, edge_f, Wn_r, bn_r, We_r, be_r, ns, es);
  select_kernel<<<2, 256, 0, stream>>>(ns, es, thr);
  nmask_kernel<<<NN / 256, 256, 0, stream>>>(ns, thr, nmask);
  weighted_kernel<<<(NE * D) / 256, 256, 0, stream>>>(edge_f, es, thr, nmask, ei, wgt);
  packw_kernel<<<(5 * 16384 + 2048 + 128 + 255) / 256, 256, 0, stream>>>(Wq, Wk, Wv, Wo, W1, W2, bq,
                                                                          bpack, bqs);
  // fused q/k/v projections (Wq & bq pre-scaled by 1/sqrt(dh)); activations read once
  qkv_gemm_kernel<<<512, 256, 0, stream>>>(wgt, bpack, bqs, bk, bv, q_h, k_h, v_h);
  vpack_kernel<<<(4 * 256 * 2 * 32 * 16) / 256, 256, 0, stream>>>(v_h, vpk);
  // flash attention over adjacency mask: 512 q-tiles x 4 heads = 2048 waves
  attn_kernel<<<256, 256, 0, stream>>>(q_h, k_h, vpk, ei, attn);
  // output projection + MLP
  gemm_kernel<<<512, 256, 0, stream>>>(attn, bpack + 3 * 16384, bo, 128, 8, 0, o_h, nullptr);
  gemm_kernel<<<512, 256, 0, stream>>>(o_h, bpack + 4 * 16384, b1, 128, 8, 1, h1, nullptr);
  gemm_kernel<<<64, 256, 0, stream>>>(h1, bpack + 5 * 16384, b2, 10, 1, 2, nullptr, logits);
}
